// STICKYLlamaAttention_45354854645977
// MI455X (gfx1250) — matrix operand order repaired and checked by reading, not compiled, not run on hardware
//
#include <hip/hip_runtime.h>
#include <hip/hip_bf16.h>
#include <math.h>

// ---------------------------------------------------------------------------
// Llama attention layer for MI455X (gfx1250, wave32, WMMA + TDM).
// All GEMM stages routed through v_wmma_f32_16x16x32_f16 (fp32 accum).
// LDS tile staging via tensor_load_to_lds (TENSORcnt) where layout is
// row-major; transposed tiles staged manually.
// S=2048, H=4096, 32 Q heads / 8 KV heads, hd=128 (GQA group = 4).
// Workspace: ~143 MB of f16 staging (casts + Q/K/V + attn out).
// ---------------------------------------------------------------------------

typedef __attribute__((ext_vector_type(16))) _Float16 v16h;
typedef __attribute__((ext_vector_type(8)))  _Float16 h8;
typedef __attribute__((ext_vector_type(8)))  float    v8f;
typedef __attribute__((ext_vector_type(4)))  unsigned u32x4;
typedef __attribute__((ext_vector_type(8)))  int      i32x8;
typedef __attribute__((ext_vector_type(4)))  int      i32x4;

#define S_LEN 2048
#define HID   4096
#define NH    32
#define NKV   8
#define HD    128

// ---------------------------------------------------------------------------
// TDM: DMA a 2D f16 tile (tile_w elems x tile_h rows, row stride in elems)
// from global memory into LDS, with optional row padding so rows land on a
// bank-friendly stride. pad_interval_code: rows of 2^(c+1) DWORDs;
// pad_amount_code: (c+1) DWORDs of LDS padding after each interval.
// Issue from ONE wave; complete with s_wait_tensorcnt + workgroup barrier.
// This toolchain's builtin takes 6 args: (g0, g1, g2, g3, g_ext, cpol);
// groups 2/3/ext are only read for >=3D tensors / gather, so zero-fill.
// ---------------------------------------------------------------------------
__device__ __forceinline__ void tdm_load_tile_f16(
    const _Float16* gptr, void* lds_ptr,
    unsigned tile_w, unsigned tile_h, unsigned row_stride_elems,
    unsigned pad_interval_code, unsigned pad_amount_code) {
  unsigned long long ga = (unsigned long long)(uintptr_t)gptr;
  unsigned lds_off = (unsigned)(uintptr_t)lds_ptr;  // low 32 bits = LDS byte addr

  u32x4 g0;
  g0[0] = 1u;                                       // count=1, user descriptor
  g0[1] = lds_off;                                  // lds_addr
  g0[2] = (unsigned)ga;                             // global_addr[31:0]
  g0[3] = (unsigned)((ga >> 32) & 0x01ffffffu)      // global_addr[56:32]
        | (2u << 30);                               // type = 2 ("image")

  unsigned long long st0 = row_stride_elems;        // dim0 stride, data_size units
  i32x8 g1;
  g1[0] = (int)((1u << 16)                          // data_size = 1 -> 2 bytes
        | (1u << 20)                                // pad_enable
        | (pad_interval_code << 22)
        | (pad_amount_code << 25));
  g1[1] = (int)((tile_w & 0xffffu) << 16);          // tensor_dim0[15:0] (=tile_w)
  g1[2] = (int)(((tile_w >> 16) & 0xffffu));        // tensor_dim0[31:16] | dim1 lo(0)
  g1[3] = (int)(0x10u                               // tensor_dim1[31:16] -> dim1=1<<20
        | ((tile_w & 0xffffu) << 16));              // tile_dim0
  g1[4] = (int)(tile_h & 0xffffu);                  // tile_dim1 (tile_dim2 = 0)
  g1[5] = (int)(st0 & 0xffffffffu);                 // tensor_dim0_stride[31:0]
  g1[6] = (int)((st0 >> 32) & 0xffffu);             // stride[47:32] | dim1_stride lo
  g1[7] = 0;
  i32x4 z4 = {0, 0, 0, 0};
  i32x8 z8 = {0, 0, 0, 0, 0, 0, 0, 0};
  __builtin_amdgcn_tensor_load_to_lds(g0, g1, z4, z4, z8, 0);
}

// ---------------------------------------------------------------------------
// fp32 -> f16 cast (grid-stride)
// ---------------------------------------------------------------------------
__global__ void cast_f32_f16(const float* __restrict__ src,
                             _Float16* __restrict__ dst, int n) {
  int i = blockIdx.x * blockDim.x + threadIdx.x;
  int stride = gridDim.x * blockDim.x;
  for (; i < n; i += stride) dst[i] = (_Float16)src[i];
}

// ---------------------------------------------------------------------------
// WMMA GEMM: C[M,N] = A[M,K] * B[K,N], f16 in, fp32 accumulate.
// Block = 128 threads (4 waves), 128x64 output tile, K-step 32.
// Wave w computes rows [w*32, w*32+32) x 64 cols: 8 WMMA accumulators,
// each B fragment reused for two A fragments.
// A tile DMA'd via TDM (padded rows); B tile staged transposed by the waves.
// ---------------------------------------------------------------------------
template <bool F16OUT>
__global__ __launch_bounds__(128) void gemm_wmma(
    const _Float16* __restrict__ A, const _Float16* __restrict__ B,
    void* __restrict__ Cout, int M, int N, int K) {
  __shared__ _Float16 As[128 * 48];   // 128 rows x 32 cols, stride 48 (96B rows)
  __shared__ _Float16 Bts[64 * 48];   // B^T: 64 cols(n) x 32 rows(k), stride 48

  const int tid  = threadIdx.x;
  const int wave = tid >> 5;
  const int lane = tid & 31;
  const int m0 = blockIdx.y * 128;
  const int n0 = blockIdx.x * 64;

  v8f acc[2][4] = {};

  for (int k0 = 0; k0 < K; k0 += 32) {
    __syncthreads();
    // A tile 128x32 via TDM: rows of 16 DWORDs (code 3), pad 8 DWORDs (code 7)
    if (wave == 0)
      tdm_load_tile_f16(A + (size_t)m0 * K + k0, As, 32, 128, (unsigned)K, 3, 7);
    // B tile 32x64, stored transposed, by all threads (overlaps the DMA)
    for (int it = 0; it < 2; ++it) {
      int v = tid + it * 128;
      int r = v >> 3;                  // 8 h8 per 64-wide row
      int c = (v & 7) * 8;
      h8 d = *(const h8*)(B + (size_t)(k0 + r) * N + n0 + c);
      for (int e = 0; e < 8; ++e) Bts[(c + e) * 48 + r] = d[e];
    }
    if (wave == 0) __builtin_amdgcn_s_wait_tensorcnt(0);
    __syncthreads();

    // A fragments (16x32): lane row m = lane&15; e<8 -> k = hi*8+e,
    // e>=8 -> k = 16+hi*8+(e-8), hi = lane>=16.
    v16h af[2];
    for (int mt = 0; mt < 2; ++mt) {
      int m  = wave * 32 + mt * 16 + (lane & 15);
      int hi = (lane >> 4) * 8;
      h8 lo = *(const h8*)(As + m * 48 + hi);
      h8 up = *(const h8*)(As + m * 48 + 16 + hi);
      for (int e = 0; e < 8; ++e) { af[mt][e] = lo[e]; af[mt][8 + e] = up[e]; }
    }
    // B fragments (32x16): lane col n = lane&15, same k order as A fragments.
    for (int nt = 0; nt < 4; ++nt) {
      int n  = nt * 16 + (lane & 15);
      int kb = (lane >> 4) * 8;
      h8 lo = *(const h8*)(Bts + n * 48 + kb);
      h8 up = *(const h8*)(Bts + n * 48 + kb + 16);
      v16h b;
      for (int e = 0; e < 8; ++e) { b[e] = lo[e]; b[8 + e] = up[e]; }
      acc[0][nt] = __builtin_amdgcn_wmma_f32_16x16x32_f16(
          false, af[0], false, b, (short)0, acc[0][nt], false, false);
      acc[1][nt] = __builtin_amdgcn_wmma_f32_16x16x32_f16(
          false, af[1], false, b, (short)0, acc[1][nt], false, false);
    }
  }

  // C layout: lane l element r -> row (l>=16?8:0)+r, col l&15 within 16x16 tile
  const int ncol0 = n0 + (lane & 15);
  for (int mt = 0; mt < 2; ++mt) {
    const int mrow = m0 + wave * 32 + mt * 16 + ((lane >> 4) << 3);
    for (int nt = 0; nt < 4; ++nt) {
      for (int r = 0; r < 8; ++r) {
        size_t idx = (size_t)(mrow + r) * N + (ncol0 + nt * 16);
        if (F16OUT) ((_Float16*)Cout)[idx] = (_Float16)acc[mt][nt][r];
        else        ((float*)Cout)[idx]    = acc[mt][nt][r];
      }
    }
  }
}

// ---------------------------------------------------------------------------
// RoPE in place on [S, nheads*128] f16; pair (t, t+64), theta=10000.
// ---------------------------------------------------------------------------
__global__ void rope_kernel(_Float16* __restrict__ X, int nheads) {
  const int s = blockIdx.x, head = blockIdx.y, t = threadIdx.x;  // t in [0,64)
  size_t base = (size_t)s * nheads * HD + (size_t)head * HD;
  float inv_freq = __expf(-__logf(10000.0f) * (float)t / 64.0f);
  float sn, cs;
  __sincosf((float)s * inv_freq, &sn, &cs);
  float x0 = (float)X[base + t];
  float x1 = (float)X[base + 64 + t];
  X[base + t]      = (_Float16)(x0 * cs - x1 * sn);
  X[base + 64 + t] = (_Float16)(x1 * cs + x0 * sn);
}

// ---------------------------------------------------------------------------
// Flash attention (causal, GQA 4:1).
// Grid: (S/64 query blocks, NH heads). Block: 128 threads = 4 waves.
// Q/K tiles DMA'd by TDM into padded-row LDS; V staged transposed by waves
// (overlapping the K DMA). Wave w owns 16 query rows: 16 WMMAs for Q*K^T,
// online softmax (shfl_xor row reductions), P via per-wave LDS slab,
// 16 WMMAs for P*V.
// ---------------------------------------------------------------------------
__global__ __launch_bounds__(128) void flash_attn(
    const _Float16* __restrict__ Q,   // [S, NH*HD]
    const _Float16* __restrict__ K,   // [S, NKV*HD]
    const _Float16* __restrict__ V,   // [S, NKV*HD]
    _Float16* __restrict__ O) {       // [S, NH*HD]
  __shared__ _Float16 Qs[64 * 136];   // 64 x 128, stride 136 (272B rows)
  __shared__ _Float16 Ks[64 * 136];
  __shared__ _Float16 Vt[128 * 72];   // V^T: 128(hd) x 64(key), stride 72
  __shared__ _Float16 Ps[4 * 16 * 72];// per-wave 16x64 P slab, stride 72

  const int tid  = threadIdx.x;
  const int wave = tid >> 5;
  const int lane = tid & 31;
  const int h    = blockIdx.y;
  const int kvh  = h >> 2;            // 32 Q heads -> 8 KV heads
  const int q0   = blockIdx.x * 64;
  const float scale = 0.08838834764831845f;  // 1/sqrt(128)

  // Q tile (64x128) via TDM: rows of 64 DWORDs (code 5), pad 4 DWORDs (code 3)
  if (wave == 0) {
    tdm_load_tile_f16(Q + (size_t)q0 * (NH * HD) + (size_t)h * HD,
                      Qs, 128, 64, NH * HD, 5, 3);
    __builtin_amdgcn_s_wait_tensorcnt(0);
  }
  __syncthreads();

  // Q A-fragments, one per 32-wide chunk of head_dim, resident for the loop.
  v16h qa[4];
  {
    int m  = wave * 16 + (lane & 15);
    int hi = (lane >> 4) * 8;
    for (int kc = 0; kc < 4; ++kc) {
      h8 lo = *(const h8*)(Qs + m * 136 + kc * 32 + hi);
      h8 up = *(const h8*)(Qs + m * 136 + kc * 32 + 16 + hi);
      for (int e = 0; e < 8; ++e) { qa[kc][e] = lo[e]; qa[kc][8 + e] = up[e]; }
    }
  }

  v8f o_acc[8] = {};
  float m_i[8], l_i[8];
  for (int r = 0; r < 8; ++r) { m_i[r] = -3.0e38f; l_i[r] = 0.f; }

  const int jmax = blockIdx.x;
  for (int j = 0; j <= jmax; ++j) {
    __syncthreads();
    // K block via TDM; V block transposed by all threads while the DMA runs.
    if (wave == 0)
      tdm_load_tile_f16(K + (size_t)(j * 64) * (NKV * HD) + (size_t)kvh * HD,
                        Ks, 128, 64, NKV * HD, 5, 3);
    for (int it = 0; it < 8; ++it) {
      int v = tid + it * 128;
      int r = v >> 4;
      int c = (v & 15) * 8;
      h8 dv = *(const h8*)(V + (size_t)(j * 64 + r) * (NKV * HD) +
                           (size_t)kvh * HD + c);
      for (int e = 0; e < 8; ++e) Vt[(c + e) * 72 + r] = dv[e];
    }
    if (j < jmax) {  // prefetch next K/V block into cache (global_prefetch)
      size_t g = (size_t)((j + 1) * 64 + (tid >> 1)) * (NKV * HD) +
                 (size_t)kvh * HD + (size_t)(tid & 1) * 64;
      __builtin_prefetch(K + g, 0, 0);
      __builtin_prefetch(V + g, 0, 0);
    }
    if (wave == 0) __builtin_amdgcn_s_wait_tensorcnt(0);
    __syncthreads();

    // ---- scores: S = Q * K^T  (16 queries x 64 keys per wave) ----
    v8f s_acc[4] = {};
    for (int kc = 0; kc < 4; ++kc) {
      for (int nt = 0; nt < 4; ++nt) {
        int key = nt * 16 + (lane & 15);
        int kb  = kc * 32 + (lane >> 4) * 8;
        h8 lo = *(const h8*)(Ks + key * 136 + kb);
        h8 up = *(const h8*)(Ks + key * 136 + kb + 16);
        v16h b;
        for (int e = 0; e < 8; ++e) { b[e] = lo[e]; b[8 + e] = up[e]; }
        s_acc[nt] = __builtin_amdgcn_wmma_f32_16x16x32_f16(
            false, qa[kc], false, b, (short)0, s_acc[nt], false, false);
      }
    }

    // ---- scale + causal mask (only bites on the diagonal block) ----
    const int qrow = q0 + wave * 16 + ((lane >> 4) << 3);
    for (int nt = 0; nt < 4; ++nt) {
      int key = j * 64 + nt * 16 + (lane & 15);
      for (int r = 0; r < 8; ++r) {
        float sv = s_acc[nt][r] * scale;
        if (key > qrow + r) sv = -3.0e38f;
        s_acc[nt][r] = sv;
      }
    }

    // ---- online softmax: row max / exp / row sum ----
    float rmax[8];
    for (int r = 0; r < 8; ++r)
      rmax[r] = fmaxf(fmaxf(s_acc[0][r], s_acc[1][r]),
                      fmaxf(s_acc[2][r], s_acc[3][r]));
    for (int off = 1; off < 16; off <<= 1)
      for (int r = 0; r < 8; ++r)
        rmax[r] = fmaxf(rmax[r], __shfl_xor(rmax[r], off, 16));

    float corr[8], rsum[8];
    for (int r = 0; r < 8; ++r) {
      float mn = fmaxf(m_i[r], rmax[r]);
      corr[r] = __expf(m_i[r] - mn);
      m_i[r] = mn;
      rsum[r] = 0.f;
    }

    _Float16* pw = Ps + wave * (16 * 72);
    for (int nt = 0; nt < 4; ++nt) {
      for (int r = 0; r < 8; ++r) {
        float p = __expf(s_acc[nt][r] - m_i[r]);
        rsum[r] += p;
        pw[(((lane >> 4) << 3) + r) * 72 + nt * 16 + (lane & 15)] = (_Float16)p;
      }
    }
    for (int off = 1; off < 16; off <<= 1)
      for (int r = 0; r < 8; ++r)
        rsum[r] += __shfl_xor(rsum[r], off, 16);
    for (int r = 0; r < 8; ++r) l_i[r] = l_i[r] * corr[r] + rsum[r];
    for (int nt = 0; nt < 8; ++nt)
      for (int r = 0; r < 8; ++r)
        o_acc[nt][r] *= corr[r];

    // Per-wave LDS slab: drain our ds stores before fragment reads.
    asm volatile("s_wait_dscnt 0" ::: "memory");

    // ---- O += P * V ----
    for (int kc = 0; kc < 2; ++kc) {
      v16h pa;
      {
        int m  = lane & 15;
        int hi = (lane >> 4) * 8;
        h8 lo = *(const h8*)(pw + m * 72 + kc * 32 + hi);
        h8 up = *(const h8*)(pw + m * 72 + kc * 32 + 16 + hi);
        for (int e = 0; e < 8; ++e) { pa[e] = lo[e]; pa[8 + e] = up[e]; }
      }
      for (int nt = 0; nt < 8; ++nt) {
        int n  = nt * 16 + (lane & 15);
        int kb = kc * 32 + (lane >> 4) * 8;
        h8 lo = *(const h8*)(Vt + n * 72 + kb);
        h8 up = *(const h8*)(Vt + n * 72 + kb + 16);
        v16h b;
        for (int e = 0; e < 8; ++e) { b[e] = lo[e]; b[8 + e] = up[e]; }
        o_acc[nt] = __builtin_amdgcn_wmma_f32_16x16x32_f16(
            false, pa, false, b, (short)0, o_acc[nt], false, false);
      }
    }
  }

  // ---- normalize and store ----
  const int mrow = q0 + wave * 16 + ((lane >> 4) << 3);
  for (int r = 0; r < 8; ++r) {
    float inv = 1.0f / l_i[r];
    for (int nt = 0; nt < 8; ++nt) {
      O[(size_t)(mrow + r) * (NH * HD) + (size_t)h * HD + nt * 16 + (lane & 15)] =
          (_Float16)(o_acc[nt][r] * inv);
    }
  }
}

// ---------------------------------------------------------------------------
// Host-side launch: cast -> QKV GEMMs -> RoPE -> flash attention -> O GEMM.
// ---------------------------------------------------------------------------
extern "C" void kernel_launch(void* const* d_in, const int* in_sizes, int n_in,
                              void* d_out, int out_size, void* d_ws, size_t ws_size,
                              hipStream_t stream) {
  const float* hs = (const float*)d_in[0];
  const float* Wq = (const float*)d_in[1];
  const float* Wk = (const float*)d_in[2];
  const float* Wv = (const float*)d_in[3];
  const float* Wo = (const float*)d_in[4];
  float* out = (float*)d_out;

  _Float16* ws = (_Float16*)d_ws;
  size_t off = 0;
  _Float16* Xh  = ws + off; off += (size_t)S_LEN * HID;
  _Float16* Wqh = ws + off; off += (size_t)HID * (NH * HD);
  _Float16* Wkh = ws + off; off += (size_t)HID * (NKV * HD);
  _Float16* Wvh = ws + off; off += (size_t)HID * (NKV * HD);
  _Float16* Woh = ws + off; off += (size_t)(NH * HD) * HID;
  _Float16* Qp  = ws + off; off += (size_t)S_LEN * (NH * HD);
  _Float16* Kp  = ws + off; off += (size_t)S_LEN * (NKV * HD);
  _Float16* Vp  = ws + off; off += (size_t)S_LEN * (NKV * HD);
  _Float16* Ob  = ws + off; off += (size_t)S_LEN * (NH * HD);
  (void)ws_size; (void)in_sizes; (void)n_in; (void)out_size;

  auto cast = [&](const float* s, _Float16* d, size_t n) {
    int blocks = (int)((n + 255) / 256);
    if (blocks > 4096) blocks = 4096;
    cast_f32_f16<<<blocks, 256, 0, stream>>>(s, d, (int)n);
  };
  cast(hs, Xh, (size_t)S_LEN * HID);
  cast(Wq, Wqh, (size_t)HID * NH * HD);
  cast(Wk, Wkh, (size_t)HID * NKV * HD);
  cast(Wv, Wvh, (size_t)HID * NKV * HD);
  cast(Wo, Woh, (size_t)NH * HD * HID);

  dim3 blk(128);
  // Q/K/V projections (f16 out)
  gemm_wmma<true><<<dim3((NH * HD) / 64,  S_LEN / 128), blk, 0, stream>>>(
      Xh, Wqh, Qp, S_LEN, NH * HD, HID);
  gemm_wmma<true><<<dim3((NKV * HD) / 64, S_LEN / 128), blk, 0, stream>>>(
      Xh, Wkh, Kp, S_LEN, NKV * HD, HID);
  gemm_wmma<true><<<dim3((NKV * HD) / 64, S_LEN / 128), blk, 0, stream>>>(
      Xh, Wvh, Vp, S_LEN, NKV * HD, HID);

  // RoPE on Q and K
  rope_kernel<<<dim3(S_LEN, NH),  64, 0, stream>>>(Qp, NH);
  rope_kernel<<<dim3(S_LEN, NKV), 64, 0, stream>>>(Kp, NKV);

  // causal GQA flash attention
  flash_attn<<<dim3(S_LEN / 64, NH), blk, 0, stream>>>(Qp, Kp, Vp, Ob);

  // output projection (fp32 out -> d_out)
  gemm_wmma<false><<<dim3(HID / 64, S_LEN / 128), blk, 0, stream>>>(
      Ob, Woh, out, S_LEN, HID, NH * HD);
}
